// SupHeads_88648124991475
// MI455X (gfx1250) — matrix operand order repaired and checked by reading, not compiled, not run on hardware
//
#include <hip/hip_runtime.h>
#include <hip/hip_bf16.h>

// ---------------- problem constants (from reference) ----------------
#define TN 4
#define SS 6
#define DD 512
#define HH 4
#define DHD 128
#define PP 3072      // S*D
#define MMDIM 6144   // 2*P
#define CC 1000

// ---------------- WMMA types ----------------
typedef __attribute__((ext_vector_type(16))) __bf16 v16bf;
typedef __attribute__((ext_vector_type(8)))  float  v8f;

// =====================================================================
// Stage 1: LN_in + K/V projection + 6-wide softmax attention -> ctx
// one 256-thread block per (t, b)
// =====================================================================
__global__ void attn_mixer_ctx(const float* __restrict__ feats,
                               const float* __restrict__ g_in,
                               const float* __restrict__ b_in,
                               const float* __restrict__ Wk,
                               const float* __restrict__ Wv,
                               const float* __restrict__ q,
                               float* __restrict__ ctx, int Brows)
{
    const int b = blockIdx.x, t = blockIdx.y, tid = threadIdx.x;
    __shared__ float xs[SS][DD];
    __shared__ float Ks[SS][DD];
    __shared__ float Vs[SS][DD];
    __shared__ float red1[256], red2[256];
    __shared__ float mean_s[SS], rstd_s[SS];
    __shared__ float sc[HH][SS], alpha[HH][SS];

    const float* xrow = feats + (size_t)b * (SS * DD);
    for (int i = tid; i < SS * DD; i += 256) xs[i >> 9][i & 511] = xrow[i];
    __syncthreads();

    // per-scale mean / rstd over D=512
    for (int s = 0; s < SS; ++s) {
        float a = xs[s][tid], c = xs[s][tid + 256];
        red1[tid] = a + c; red2[tid] = a * a + c * c;
        __syncthreads();
        for (int off = 128; off > 0; off >>= 1) {
            if (tid < off) { red1[tid] += red1[tid + off]; red2[tid] += red2[tid + off]; }
            __syncthreads();
        }
        if (tid == 0) {
            float m = red1[0] * (1.0f / DD);
            float v = red2[0] * (1.0f / DD) - m * m;
            mean_s[s] = m; rstd_s[s] = rsqrtf(v + 1e-5f);
        }
        __syncthreads();
    }
    // normalize with task affine
    for (int i = tid; i < SS * DD; i += 256) {
        int s = i >> 9, d = i & 511;
        xs[s][d] = (xs[s][d] - mean_s[s]) * rstd_s[s] * g_in[t * DD + d] + b_in[t * DD + d];
    }
    __syncthreads();

    // K / V projections (6x512 out each, K=512 dot)
    for (int i = tid; i < SS * DD; i += 256) {
        int s = i >> 9, k = i & 511;
        const float* wk = Wk + ((size_t)t * DD + k) * DD;
        const float* wv = Wv + ((size_t)t * DD + k) * DD;
        float ak = 0.f, av = 0.f;
        for (int d = 0; d < DD; ++d) {
            float xv = xs[s][d];
            ak = fmaf(xv, wk[d], ak);
            av = fmaf(xv, wv[d], av);
        }
        Ks[s][k] = ak; Vs[s][k] = av;
    }
    __syncthreads();

    // scores[h][s] = <K[s, h*dh:], q[t,h]> / sqrt(dh)
    if (tid < HH * SS) {
        int h = tid / SS, s = tid % SS;
        const float* qh = q + ((size_t)t * HH + h) * DHD;
        const float* kr = &Ks[s][h * DHD];
        float a = 0.f;
        for (int d = 0; d < DHD; ++d) a = fmaf(kr[d], qh[d], a);
        sc[h][s] = a * 0.08838834764831845f;   // 1/sqrt(128)
    }
    __syncthreads();
    // softmax over S=6
    if (tid < HH) {
        float mx = sc[tid][0];
        for (int s = 1; s < SS; ++s) mx = fmaxf(mx, sc[tid][s]);
        float e[SS], sum = 0.f;
        for (int s = 0; s < SS; ++s) { e[s] = __expf(sc[tid][s] - mx); sum += e[s]; }
        float inv = 1.f / sum;
        for (int s = 0; s < SS; ++s) alpha[tid][s] = e[s] * inv;
    }
    __syncthreads();
    // ctx[k] = sum_s alpha[h][s] * V[s][k]
    float* crow = ctx + ((size_t)t * Brows + b) * (HH * DHD);
    for (int k = tid; k < HH * DHD; k += 256) {
        int h = k >> 7;
        float a = 0.f;
        for (int s = 0; s < SS; ++s) a = fmaf(alpha[h][s], Vs[s][k], a);
        crow[k] = a;
    }
}

// =====================================================================
// Task-batched WMMA GEMM:  C[t] = A[t] (256 x K) * W[t]^T (N x K) + bias
// (optional ReLU). bf16 operands, f32 accumulate.
//
// 512 threads = 16 waves; block tile = 256 rows x 64 cols so each fp32
// weight tile streams from HBM exactly once per task. The 64x32 weight
// tile is copied global->LDS with async DMA (double buffered,
// global_load_async_to_lds_b128 + s_wait_asynccnt), converted once with
// hardware v_cvt_pk_bf16_f32 into a shared bf16 tile, and all 16 waves
// pull B fragments from LDS. Last K-step is peeled so the steady-state
// loop is branch-free.
// grid = (ceil(N/64), 1, TN)   -- requires Brows == 256
// =====================================================================
template<bool RELU>
__global__ void __launch_bounds__(512)
wmma_gemm_bias(const float* __restrict__ Aall,
               const float* __restrict__ Wall,
               const float* __restrict__ biasall,
               float* __restrict__ Call,
               int Brows, int N, int K)
{
    const int t = blockIdx.z;
    const float* A    = Aall    + (size_t)t * Brows * K;
    const float* W    = Wall    + (size_t)t * N * K;
    const float* bias = biasall + (size_t)t * N;
    float*       Cm   = Call    + (size_t)t * Brows * N;

    __shared__ float  Wf32[2][64 * 32];   // fp32 staging (async DMA target)
    __shared__ __bf16 Wbf[64][32];        // shared bf16 weight tile

    const int tid   = threadIdx.x;
    const int lane  = tid & 31;
    const int wave  = tid >> 5;           // 0..15
    const int m0    = wave * 16;          // 16 rows per wave -> 256 rows
    const int nbase = blockIdx.x * 64;

    // async-copy assignment: each thread moves one 16B chunk per tile
    const int  flat   = tid * 4;          // float index within 64x32 tile
    const int  wrow   = flat >> 5;        // 0..63  (local n)
    const int  wcol   = flat & 31;        // 0..31  (local k)
    const bool wvalid = (nbase + wrow) < N;
    const unsigned ldsA = (unsigned)(size_t)&Wf32[0][flat];
    const unsigned ldsB = (unsigned)(size_t)&Wf32[1][flat];
    const float* gsrc = W + (size_t)(nbase + wrow) * K + wcol;

    // fragment geometry (contiguous k = 16*(lane>>4) + s, same for A and B)
    const int ar  = lane & 15;
    const int akb = (lane >> 4) << 4;     // A: K base 0 / 16
    const int bn  = lane & 15;
    const int bkb = (lane >> 4) << 4;     // B: K base 0 / 16

    v8f acc[4] = {};
    const int ksteps = K >> 5;            // >= 2 for all our shapes

    // per-K-step body: convert staged tile, build fragments, 4 WMMAs
    auto body = [&](int i) {
        __syncthreads();
        // one cooperative fp32 -> bf16 conversion of the tile
        {
            const float* sf = &Wf32[i & 1][flat];
            __bf16* db = &Wbf[wrow][wcol];
            if (wvalid) {
                db[0] = (__bf16)sf[0]; db[1] = (__bf16)sf[1];
                db[2] = (__bf16)sf[2]; db[3] = (__bf16)sf[3];
            } else {
                db[0] = (__bf16)0.f; db[1] = (__bf16)0.f;
                db[2] = (__bf16)0.f; db[3] = (__bf16)0.f;
            }
        }
        __syncthreads();

        // A fragment: per-wave rows, global fp32 -> bf16 regs (hw cvt)
        const int k0 = i << 5;
        v16bf a;
        const float* ap = A + (size_t)(m0 + ar) * K + k0 + akb;
#pragma unroll
        for (int j = 0; j < 8; ++j) {
            a[j]     = (__bf16)ap[j];
            a[8 + j] = (__bf16)ap[8 + j];
        }

        // load ALL 4 B fragments first, then 4 back-to-back WMMAs
        v16bf bm[4];
#pragma unroll
        for (int j = 0; j < 4; ++j) {
            const __bf16* bp = &Wbf[j * 16 + bn][bkb];
#pragma unroll
            for (int ii = 0; ii < 16; ++ii) bm[j][ii] = bp[ii];
        }
#pragma unroll
        for (int j = 0; j < 4; ++j) {
            acc[j] = __builtin_amdgcn_wmma_f32_16x16x32_bf16(
                false, a, false, bm[j], (short)0, acc[j], false, false);
        }
    };

    // prologue: DMA tile 0 into buffer 0
    if (wvalid) {
        asm volatile("global_load_async_to_lds_b128 %0, %1, off"
                     :: "v"(ldsA), "v"(gsrc) : "memory");
    }

    // steady state: always prefetch tile i+1, wait for tile i, compute
    for (int i = 0; i < ksteps - 1; ++i) {
        const unsigned dst = ((i + 1) & 1) ? ldsB : ldsA;
        const float* src = gsrc + ((i + 1) << 5);
        if (wvalid) {
            asm volatile("global_load_async_to_lds_b128 %0, %1, off"
                         :: "v"(dst), "v"(src) : "memory");
        }
        asm volatile("s_wait_asynccnt 0x1" ::: "memory");
        body(i);
    }
    // peeled last step: just drain the async queue and compute
    asm volatile("s_wait_asynccnt 0x0" ::: "memory");
    body(ksteps - 1);

    // C layout: VGPR v, lane -> row m0 + v + 8*(lane>>4), col = lane%16
    const int rbase = m0 + ((lane >> 4) << 3);
#pragma unroll
    for (int j = 0; j < 4; ++j) {
        int n = nbase + j * 16 + (lane & 15);
        if (n < N) {
            float bv = bias[n];
#pragma unroll
            for (int v = 0; v < 8; ++v) {
                float val = acc[j][v] + bv;
                if (RELU) val = fmaxf(val, 0.f);
                Cm[(size_t)(rbase + v) * N + n] = val;
            }
        }
    }
}

// =====================================================================
// Fused double LayerNorm over P=3072: y = LN(LN(h; g1,b1); g2,b2)
// one 256-thread block per (t,b) row
// =====================================================================
__global__ void double_ln(const float* __restrict__ h,
                          const float* __restrict__ g1, const float* __restrict__ b1,
                          const float* __restrict__ g2, const float* __restrict__ b2,
                          float* __restrict__ y, int Brows)
{
    const int row = blockIdx.x;          // t*Brows + b
    const int t   = row / Brows;
    const int tid = threadIdx.x;
    __shared__ float red1[256], red2[256];
    __shared__ float stat[2];

    const float* hr = h + (size_t)row * PP;
    float v[PP / 256];
    float s1 = 0.f, s2 = 0.f;
#pragma unroll
    for (int j = 0; j < PP / 256; ++j) {
        float x = hr[tid + j * 256];
        v[j] = x; s1 += x; s2 += x * x;
    }
    red1[tid] = s1; red2[tid] = s2; __syncthreads();
    for (int off = 128; off > 0; off >>= 1) {
        if (tid < off) { red1[tid] += red1[tid + off]; red2[tid] += red2[tid + off]; }
        __syncthreads();
    }
    if (tid == 0) {
        float m = red1[0] * (1.0f / PP);
        float var = red2[0] * (1.0f / PP) - m * m;
        stat[0] = m; stat[1] = rsqrtf(var + 1e-5f);
    }
    __syncthreads();
    float m = stat[0], r = stat[1];
    s1 = 0.f; s2 = 0.f;
#pragma unroll
    for (int j = 0; j < PP / 256; ++j) {
        int d = tid + j * 256;
        float o = (v[j] - m) * r * g1[t * PP + d] + b1[t * PP + d];
        v[j] = o; s1 += o; s2 += o * o;
    }
    __syncthreads();
    red1[tid] = s1; red2[tid] = s2; __syncthreads();
    for (int off = 128; off > 0; off >>= 1) {
        if (tid < off) { red1[tid] += red1[tid + off]; red2[tid] += red2[tid + off]; }
        __syncthreads();
    }
    if (tid == 0) {
        float mm = red1[0] * (1.0f / PP);
        float var = red2[0] * (1.0f / PP) - mm * mm;
        stat[0] = mm; stat[1] = rsqrtf(var + 1e-5f);
    }
    __syncthreads();
    m = stat[0]; r = stat[1];
    float* yr = y + (size_t)row * PP;
#pragma unroll
    for (int j = 0; j < PP / 256; ++j) {
        int d = tid + j * 256;
        yr[d] = (v[j] - m) * r * g2[t * PP + d] + b2[t * PP + d];
    }
}

// =====================================================================
extern "C" void kernel_launch(void* const* d_in, const int* in_sizes, int n_in,
                              void* d_out, int out_size, void* d_ws, size_t ws_size,
                              hipStream_t stream) {
    const float* feats    = (const float*)d_in[0];
    const float* ln_in_g  = (const float*)d_in[1];
    const float* ln_in_b  = (const float*)d_in[2];
    const float* Wk       = (const float*)d_in[3];
    const float* Wv       = (const float*)d_in[4];
    const float* q        = (const float*)d_in[5];
    const float* Wp       = (const float*)d_in[6];
    const float* bp       = (const float*)d_in[7];
    const float* ln_out_g = (const float*)d_in[8];
    const float* ln_out_b = (const float*)d_in[9];
    const float* cln_g    = (const float*)d_in[10];
    const float* cln_b    = (const float*)d_in[11];
    const float* W1       = (const float*)d_in[12];
    const float* b1       = (const float*)d_in[13];
    const float* W2       = (const float*)d_in[14];
    const float* b2       = (const float*)d_in[15];

    const int Brows = in_sizes[0] / (SS * DD);   // 256

    // workspace: ctx (2MB) | h (12MB) | y (12MB) | h1 (24MB)  ~= 52MB
    float* ctx = (float*)d_ws;
    float* h   = ctx + (size_t)TN * Brows * (HH * DHD);
    float* y   = h   + (size_t)TN * Brows * PP;
    float* h1  = y   + (size_t)TN * Brows * PP;

    // Stage 1: attention front-end -> ctx (T,B,512)
    attn_mixer_ctx<<<dim3(Brows, TN), 256, 0, stream>>>(
        feats, ln_in_g, ln_in_b, Wk, Wv, q, ctx, Brows);

    // Stage 2: h = relu(ctx @ Wp^T + bp)   (K=512, N=3072)
    wmma_gemm_bias<true><<<dim3(PP / 64, 1, TN), 512, 0, stream>>>(
        ctx, Wp, bp, h, Brows, PP, HH * DHD);

    // Stage 3: y = cln(ln_out(h))
    double_ln<<<dim3(TN * Brows), 256, 0, stream>>>(
        h, ln_out_g, ln_out_b, cln_g, cln_b, y, Brows);

    // Stage 4: h1 = relu(y @ W1^T + b1)    (K=3072, N=6144)
    wmma_gemm_bias<true><<<dim3(MMDIM / 64, 1, TN), 512, 0, stream>>>(
        y, W1, b1, h1, Brows, MMDIM, PP);

    // Stage 5: logits = h1 @ W2^T + b2     (K=6144, N=1000, edge-guarded)
    wmma_gemm_bias<false><<<dim3((CC + 63) / 64, 1, TN), 512, 0, stream>>>(
        h1, W2, b2, (float*)d_out, Brows, CC, MMDIM);
}
